// MoRALinear_58076547777082
// MI455X (gfx1250) — matrix operand order repaired and checked by reading, not compiled, not run on hardware
//
#include <hip/hip_runtime.h>

// ---------------------------------------------------------------------------
// MoRA linear for MI455X (gfx1250, wave32, WMMA, async LDS staging).
//
//   out = x @ W^T                      (16384 x 4096) x (4096 x 4096)
//   z   = rope(concat(out, out[:248]).reshape(6,724))
//   y   = z @ A^T (724x724), sliced to 4096, out += y
//
// fp32->bf16 one-time packs; base GEMM stages tiles into double-buffered LDS
// via global_load_async_to_lds_b128 (ASYNCcnt), computes with
// v_wmma_f32_16x16x32_bf16 at 64x64 per wave (16 WMMA / K-step).
// ---------------------------------------------------------------------------

typedef __attribute__((ext_vector_type(16))) __bf16 v16bf;
typedef __attribute__((ext_vector_type(8)))  float  v8f;

#define IN_DIM  4096
#define OUT_DIM 4096
#define R_DIM   724
#define RB1     6
#define HALF_R  362
#define KPAD    736                  // adapter K (R_DIM -> multiple of 32)
#define NPAD    768                  // adapter N (R_DIM -> multiple of 64)
#define M_TOK   16384                // 4 * 4096 tokens
#define M_MORA  (M_TOK * RB1)        // 98304 adapter rows

union BF16x16 {
    v16bf v;
    unsigned short u[16];
    uint4 q[2];
};

__device__ __forceinline__ unsigned short f2bf(float f) {
    unsigned int u = __float_as_uint(f);
    u += 0x7fffu + ((u >> 16) & 1u);          // round-to-nearest-even
    return (unsigned short)(u >> 16);
}

// LDS byte offset of a __shared__-derived pointer: flat LDS aperture keeps the
// wave-relative LDS byte offset in addr[31:0] (ISA 10.2 aperture mapping).
__device__ __forceinline__ unsigned lds_off(const void* p) {
    return (unsigned)(unsigned long long)p;
}

// gfx1250 async copy: 16B global -> LDS per lane, tracked with ASYNCcnt.
__device__ __forceinline__ void async_cp16(const void* gsrc, unsigned ldsDst) {
    asm volatile("global_load_async_to_lds_b128 %0, %1, off"
                 :: "v"(ldsDst), "v"(gsrc) : "memory");
}

// ---------------------------------------------------------------------------
// Pack kernel: fp32 -> bf16, 8 elements per thread (32B read / 16B write).
// ---------------------------------------------------------------------------
__global__ void mora_pack_bf16_kernel(const float* __restrict__ src,
                                      unsigned short* __restrict__ dst,
                                      long long n8) {
    long long i = (long long)blockIdx.x * blockDim.x + threadIdx.x;
    if (i >= n8) return;
    const float4* s = (const float4*)(src + i * 8);
    float4 f0 = s[0], f1 = s[1];
    union { uint4 q; unsigned short u[8]; } o;
    o.u[0] = f2bf(f0.x); o.u[1] = f2bf(f0.y); o.u[2] = f2bf(f0.z); o.u[3] = f2bf(f0.w);
    o.u[4] = f2bf(f1.x); o.u[5] = f2bf(f1.y); o.u[6] = f2bf(f1.z); o.u[7] = f2bf(f1.w);
    ((uint4*)dst)[i] = o.q;
}

// ---------------------------------------------------------------------------
// Kernel 0: cos/sin tables, emb[n, r] = n * base^(-2*(r%362)/724)
// ---------------------------------------------------------------------------
__global__ void mora_freqs_kernel(float* __restrict__ cosT, float* __restrict__ sinT) {
    int idx = blockIdx.x * blockDim.x + threadIdx.x;
    if (idx >= RB1 * R_DIM) return;
    int n = idx / R_DIM;
    int r = idx % R_DIM;
    int j = r % HALF_R;
    float inv = __expf(-(float)(2 * j) * (9.210340371976184f / (float)R_DIM));
    float e   = (float)n * inv;
    cosT[idx] = __cosf(e);
    sinT[idx] = __sinf(e);
}

// ---------------------------------------------------------------------------
// Kernel 2b: pack lora_a (724x724 fp32) into zero-padded 768x736 bf16.
// ---------------------------------------------------------------------------
__global__ void mora_pack_a_kernel(const float* __restrict__ A,
                                   unsigned short* __restrict__ Ab) {
    int idx = blockIdx.x * blockDim.x + threadIdx.x;
    if (idx >= NPAD * KPAD) return;
    int p = idx / KPAD;
    int r = idx % KPAD;
    unsigned short v = 0;
    if (p < R_DIM && r < R_DIM) v = f2bf(A[p * R_DIM + r]);
    Ab[idx] = v;
}

// ---------------------------------------------------------------------------
// Kernel 1: out = x @ W^T via WMMA bf16, async-LDS double-buffered.
// Block = 256 threads = 8 waves (4 wrow x 2 wcol). Block tile 256M x 128N.
// Wave tile 64M x 64N = 16 WMMA per K-step of 32.
// LDS: A tile 256x32 (16KB) + B tile 128x32 (8KB), x2 buffers = 48KB.
// Tiles stored as (row, 32 K): fragment reads are ds_load_b128. All 8 frags
// (4 A + 4 B) live in distinct registers so s_wait_dscnt can stagger.
// ---------------------------------------------------------------------------
#define NK_BASE (IN_DIM / 32)        // 128 K-steps

__global__ __launch_bounds__(256)
void mora_base_gemm_kernel(const unsigned short* __restrict__ Xb,
                           const unsigned short* __restrict__ Wb,
                           float* __restrict__ Out) {
    __shared__ unsigned short As[2][256 * 32];   // 16KB per buffer
    __shared__ unsigned short Bs[2][128 * 32];   //  8KB per buffer

    const int tid  = threadIdx.x;
    const int lane = tid & 31;
    const int wave = tid >> 5;
    const int m    = lane & 15;
    const int h    = lane >> 4;                  // 0/1: K half-bank
    const int wrow = wave >> 1;                  // 0..3 -> 64-row strip
    const int wcol = wave & 1;                   // 0..1 -> 64-col strip
    const int row0 = blockIdx.y * 256;
    const int col0 = blockIdx.x * 128;

    // Cooperative staging split: thread handles 4 A-chunks + 2 B-chunks of 16B.
    const int aRow = tid >> 2;                   // 0..63 (rows, x4 strips of 64)
    const int aSeg = tid & 3;                    // 16B segment within a 64B row
    const unsigned short* gA = Xb + (size_t)(row0 + aRow) * IN_DIM + aSeg * 8;
    const unsigned short* gB = Wb + (size_t)(col0 + aRow) * IN_DIM + aSeg * 8;

    v8f acc[4][4] = {};

    // ---- stage K-step 0 into buffer 0 ----
    {
#pragma unroll
        for (int s = 0; s < 4; ++s)
            async_cp16(gA + (size_t)(s * 64) * IN_DIM,
                       lds_off(&As[0][(aRow + s * 64) * 32 + aSeg * 8]));
#pragma unroll
        for (int s = 0; s < 2; ++s)
            async_cp16(gB + (size_t)(s * 64) * IN_DIM,
                       lds_off(&Bs[0][(aRow + s * 64) * 32 + aSeg * 8]));
    }

    for (int it = 0; it < NK_BASE; ++it) {
        const int buf = it & 1;
        if (it + 1 < NK_BASE) {
            const int kn = (it + 1) * 32;
#pragma unroll
            for (int s = 0; s < 4; ++s)
                async_cp16(gA + (size_t)(s * 64) * IN_DIM + kn,
                           lds_off(&As[buf ^ 1][(aRow + s * 64) * 32 + aSeg * 8]));
#pragma unroll
            for (int s = 0; s < 2; ++s)
                async_cp16(gB + (size_t)(s * 64) * IN_DIM + kn,
                           lds_off(&Bs[buf ^ 1][(aRow + s * 64) * 32 + aSeg * 8]));
            asm volatile("s_wait_asynccnt 6" ::: "memory");   // stage `it` done
        } else {
            asm volatile("s_wait_asynccnt 0" ::: "memory");
        }
        __syncthreads();                                       // tiles visible

        // ---- compute from LDS buffer `buf`: load all frags, then 16 WMMA ----
        BF16x16 a[4], b[4];
#pragma unroll
        for (int i = 0; i < 4; ++i) {
            const int e = (wrow * 64 + 16 * i + m) * 32;       // element index
            a[i].q[0] = *(const uint4*)&As[buf][e + 16 * h];
            a[i].q[1] = *(const uint4*)&As[buf][e + 16 * h + 8];
        }
#pragma unroll
        for (int j = 0; j < 4; ++j) {
            const int e = (wcol * 64 + 16 * j + m) * 32 + 16 * h;
            const uint4* bp = (const uint4*)&Bs[buf][e];
            b[j].q[0] = bp[0];
            b[j].q[1] = bp[1];
        }
#pragma unroll
        for (int j = 0; j < 4; ++j)
#pragma unroll
            for (int i = 0; i < 4; ++i)
                acc[i][j] = __builtin_amdgcn_wmma_f32_16x16x32_bf16(
                    false, a[i].v, false, b[j].v, (short)0, acc[i][j], false, false);
        __syncthreads();                                       // done reading buf
    }

    // D layout: VGPR g, lane l -> N = colBase + (l%16), M = rowBase + 8*(l/16) + g
#pragma unroll
    for (int i = 0; i < 4; ++i) {
        const int mrb = row0 + wrow * 64 + 16 * i + 8 * h;
#pragma unroll
        for (int g = 0; g < 8; ++g) {
            float* o = Out + (size_t)(mrb + g) * OUT_DIM + col0 + wcol * 64 + m;
            o[0]  = acc[i][0][g];
            o[16] = acc[i][1][g];
            o[32] = acc[i][2][g];
            o[48] = acc[i][3][g];
        }
    }
}

// ---------------------------------------------------------------------------
// Kernel 2a: build rotated Z in bf16, padded K 724 -> 736 (zeros).
// ---------------------------------------------------------------------------
__global__ void mora_rope_pack_kernel(const float* __restrict__ Out,
                                      const float* __restrict__ cosT,
                                      const float* __restrict__ sinT,
                                      unsigned short* __restrict__ Zr) {
    long long idx = (long long)blockIdx.x * blockDim.x + threadIdx.x;
    const long long total = (long long)M_TOK * RB1 * KPAD;
    if (idx >= total) return;
    int r = (int)(idx % KPAD);
    long long tn = idx / KPAD;
    int n = (int)(tn % RB1);
    long long t = tn / RB1;

    unsigned short outv = 0;
    if (r < R_DIM) {
        int col  = n * R_DIM + r;
        int src  = (col  < OUT_DIM) ? col  : col  - OUT_DIM;
        int pr   = (r < HALF_R) ? (r + HALF_R) : (r - HALF_R);
        float sg = (r < HALF_R) ? -1.0f : 1.0f;
        int pcol = n * R_DIM + pr;
        int psrc = (pcol < OUT_DIM) ? pcol : pcol - OUT_DIM;
        float z  = Out[t * OUT_DIM + src];
        float zp = Out[t * OUT_DIM + psrc];
        float v  = z * cosT[n * R_DIM + r] + sg * zp * sinT[n * R_DIM + r];
        outv = f2bf(v);
    }
    Zr[idx] = outv;
}

// ---------------------------------------------------------------------------
// Kernel 2c: Y = Zr (98304 x 736) @ Ab^T (768 x 736) via WMMA; fused epilogue
// scatters y into Out through the (n*724 + p) < 4096 slice: Out += y.
// Block = 8 waves; wave tile 16 x 64; block 128 rows x 64 cols; grid (12, 768).
// Ab is ~1MB (L2-resident); Zr streams.
// ---------------------------------------------------------------------------
__global__ __launch_bounds__(256)
void mora_adapter_gemm_kernel(const unsigned short* __restrict__ Zr,
                              const unsigned short* __restrict__ Ab,
                              float* __restrict__ Out) {
    const int lane = threadIdx.x & 31;
    const int wave = threadIdx.x >> 5;
    const int m    = lane & 15;
    const int h    = lane >> 4;
    const int row  = blockIdx.y * 128 + wave * 16 + m;   // < 98304
    const int col0 = blockIdx.x * 64;                    // < 768

    const unsigned short* zrow  = Zr + (size_t)row * KPAD;
    const unsigned short* brow0 = Ab + (size_t)(col0 +  0 + m) * KPAD;
    const unsigned short* brow1 = Ab + (size_t)(col0 + 16 + m) * KPAD;
    const unsigned short* brow2 = Ab + (size_t)(col0 + 32 + m) * KPAD;
    const unsigned short* brow3 = Ab + (size_t)(col0 + 48 + m) * KPAD;

    v8f acc0 = {}, acc1 = {}, acc2 = {}, acc3 = {};
    for (int k0 = 0; k0 < KPAD; k0 += 32) {
        BF16x16 a, b0, b1, b2, b3;
        a.q[0] = ((const uint4*)(zrow + k0 + 16 * h))[0];
        a.q[1] = ((const uint4*)(zrow + k0 + 16 * h + 8))[0];
        const uint4* p0 = (const uint4*)(brow0 + k0 + 16 * h);
        const uint4* p1 = (const uint4*)(brow1 + k0 + 16 * h);
        const uint4* p2 = (const uint4*)(brow2 + k0 + 16 * h);
        const uint4* p3 = (const uint4*)(brow3 + k0 + 16 * h);
        b0.q[0] = p0[0]; b0.q[1] = p0[1];
        b1.q[0] = p1[0]; b1.q[1] = p1[1];
        b2.q[0] = p2[0]; b2.q[1] = p2[1];
        b3.q[0] = p3[0]; b3.q[1] = p3[1];
        acc0 = __builtin_amdgcn_wmma_f32_16x16x32_bf16(false, a.v, false, b0.v, (short)0, acc0, false, false);
        acc1 = __builtin_amdgcn_wmma_f32_16x16x32_bf16(false, a.v, false, b1.v, (short)0, acc1, false, false);
        acc2 = __builtin_amdgcn_wmma_f32_16x16x32_bf16(false, a.v, false, b2.v, (short)0, acc2, false, false);
        acc3 = __builtin_amdgcn_wmma_f32_16x16x32_bf16(false, a.v, false, b3.v, (short)0, acc3, false, false);
    }

    const int mrb = blockIdx.y * 128 + wave * 16 + 8 * h;
#pragma unroll
    for (int j = 0; j < 4; ++j) {
        v8f acc = (j == 0) ? acc0 : (j == 1) ? acc1 : (j == 2) ? acc2 : acc3;
        const int pcol = col0 + 16 * j + m;
        if (pcol < R_DIM) {
#pragma unroll
            for (int g = 0; g < 8; ++g) {
                int mr  = mrb + g;
                int t   = mr / RB1;
                int n   = mr % RB1;
                int col = n * R_DIM + pcol;
                if (col < OUT_DIM) {
                    float* o = Out + (size_t)t * OUT_DIM + col;
                    *o += acc[g];        // unique address per lane: plain RMW
                }
            }
        }
    }
}

// ---------------------------------------------------------------------------
// Launch
// ---------------------------------------------------------------------------
extern "C" void kernel_launch(void* const* d_in, const int* in_sizes, int n_in,
                              void* d_out, int out_size, void* d_ws, size_t ws_size,
                              hipStream_t stream) {
    const float* X  = (const float*)d_in[0];   // (4, 4096, 4096)
    const float* W  = (const float*)d_in[1];   // (4096, 4096)
    const float* La = (const float*)d_in[2];   // (724, 724)
    float* Out = (float*)d_out;                // (4, 4096, 4096)

    // Workspace layout (64B-aligned regions), total ~314 MB:
    //  [0        .. 17408 )   cos table (4344 f32)
    //  [17408    .. 34816 )   sin table
    //  [34816    .. 1165312)  Ab  bf16 (768*736)
    //  [1165312  .. +32MB  )  Wb  bf16 (4096*4096)
    //  [34719744 .. +128MB )  Xb  bf16 (16384*4096)
    //  [168937472.. +138MB )  Zr  bf16 (98304*736)
    char* ws = (char*)d_ws;
    float*          cosT = (float*)(ws);
    float*          sinT = (float*)(ws + 17408);
    unsigned short* Ab   = (unsigned short*)(ws + 34816);
    unsigned short* Wb   = (unsigned short*)(ws + 1165312);
    unsigned short* Xb   = (unsigned short*)(ws + 34719744);
    unsigned short* Zr   = (unsigned short*)(ws + 168937472);

    // 0) one-time packs (memory-bound) + freq tables
    {
        long long n8x = (long long)M_TOK * IN_DIM / 8;        // 8,388,608
        long long n8w = (long long)OUT_DIM * IN_DIM / 8;      // 2,097,152
        mora_pack_bf16_kernel<<<dim3((unsigned)((n8x + 255) / 256)), dim3(256), 0, stream>>>(X, Xb, n8x);
        mora_pack_bf16_kernel<<<dim3((unsigned)((n8w + 255) / 256)), dim3(256), 0, stream>>>(W, Wb, n8w);
    }
    mora_freqs_kernel<<<dim3((RB1 * R_DIM + 255) / 256), dim3(256), 0, stream>>>(cosT, sinT);
    mora_pack_a_kernel<<<dim3((NPAD * KPAD + 255) / 256), dim3(256), 0, stream>>>(La, Ab);

    // 1) base GEMM: grid (N/128, M/256) = (32, 64)
    mora_base_gemm_kernel<<<dim3(OUT_DIM / 128, M_TOK / 256), dim3(256), 0, stream>>>(Xb, Wb, Out);

    // 2a) RoPE + pack Z: 16384*6*736 elements = 282624 blocks of 256
    {
        long long total = (long long)M_TOK * RB1 * KPAD;
        int blocks = (int)((total + 255) / 256);
        mora_rope_pack_kernel<<<dim3(blocks), dim3(256), 0, stream>>>(Out, cosT, sinT, Zr);
    }

    // 2c) adapter GEMM + fused accumulate: grid (768/64, 98304/128) = (12, 768)
    mora_adapter_gemm_kernel<<<dim3(NPAD / 64, M_MORA / 128), dim3(256), 0, stream>>>(Zr, Ab, Out);
}
